// Colorizer_53395033423959
// MI455X (gfx1250) — matrix operand order repaired and checked by reading, not compiled, not run on hardware
//
#include <hip/hip_runtime.h>
#include <hip/hip_bf16.h>
#include <math.h>

// ---------------------------------------------------------------------------
// Colorizer: patch-conv features + cross-attention label propagation.
// N=8, R_T=3, T_T=1, H=W=256, C=3, PATCH=8, FEAT=256, K=16
//  feats:   [32768, 256]   (8 batches * 4 frames * 32*32 positions)
//  per n:   ref rows [n*4096, +3072), tgt rows [n*4096+3072, +1024)
// Precision: bf16 operands, fp32 WMMA accumulation, fp32 softmax state.
// ---------------------------------------------------------------------------

typedef __attribute__((ext_vector_type(8)))  __bf16 bf16x8;   // 16 bytes
typedef __attribute__((ext_vector_type(16))) __bf16 v16bf;    // 32 bytes
typedef __attribute__((ext_vector_type(8)))  float  v8f;

__device__ __forceinline__ bf16x8 ld8(const __bf16* p) {
    return *reinterpret_cast<const bf16x8*>(p);
}
__device__ __forceinline__ v16bf cat16(bf16x8 a, bf16x8 b) {
    return __builtin_shufflevector(a, b, 0,1,2,3,4,5,6,7,8,9,10,11,12,13,14,15);
}
__device__ __forceinline__ v8f wmma_bf16(v16bf a, v16bf b, v8f c) {
    return __builtin_amdgcn_wmma_f32_16x16x32_bf16(
        /*neg_a=*/false, a, /*neg_b=*/false, b,
        /*c_mod=*/(short)0, c, /*reuse_a=*/false, /*reuse_b=*/false);
}

// ---------------------------------------------------------------------------
// Kernel 0a: im2col + f32->bf16.  A[p][k], p = n*4096 + t*1024 + py*32 + px,
// k = (dy*8+dx)*3 + c.  One block per patch position, one thread per k.
// ---------------------------------------------------------------------------
__global__ __launch_bounds__(192) void k_patchify(
    const float* __restrict__ ref, const float* __restrict__ tgt,
    __bf16* __restrict__ A)
{
    int p = blockIdx.x;            // 0..32767
    int k = threadIdx.x;           // 0..191
    int n   = p >> 12;
    int t   = (p >> 10) & 3;       // 0..2 = ref frame, 3 = tgt frame
    int pix = p & 1023;
    int py  = pix >> 5, px = pix & 31;
    int dy  = k / 24;
    int rem = k - dy * 24;
    int dx  = rem / 3, c = rem - dx * 3;
    int row = py * 8 + dy, col = px * 8 + dx;
    float v;
    if (t < 3)
        v = ref[(((size_t)((n * 3 + t) * 256 + row)) * 256 + col) * 3 + c];
    else
        v = tgt[(((size_t)(n * 256 + row)) * 256 + col) * 3 + c];
    A[(size_t)p * 192 + k] = (__bf16)v;
}

// ---------------------------------------------------------------------------
// Kernel 0b: weights HWIO [192,256] f32 -> transposed bf16 Bt[f][k] (k contig)
// ---------------------------------------------------------------------------
__global__ __launch_bounds__(256) void k_wcvt(
    const float* __restrict__ w, __bf16* __restrict__ Bt)
{
    int kk = blockIdx.x;           // 0..191
    int f  = threadIdx.x;          // 0..255
    Bt[(size_t)f * 192 + kk] = (__bf16)w[(size_t)kk * 256 + f];
}

// ---------------------------------------------------------------------------
// Kernel 0c: labels [8][3072][16] f32 -> transposed bf16 LT[n][kc][r] (r contig)
// ---------------------------------------------------------------------------
__global__ __launch_bounds__(256) void k_lcvt(
    const float* __restrict__ lab, __bf16* __restrict__ LT)
{
    int idx = blockIdx.x * 256 + threadIdx.x;    // < 393216
    int n   = idx / 49152;
    int rem = idx - n * 49152;
    int r   = rem >> 4;
    int kc  = rem & 15;
    LT[((size_t)(n * 16 + kc)) * 3072 + r] = (__bf16)lab[idx];
}

// ---------------------------------------------------------------------------
// Kernel 1: feature GEMM  F[32768,256] = A[32768,192] x Bt^T, bf16 WMMA.
// One wave = one 16x16 tile, K = 192 = 6 k-steps of 32.
// ---------------------------------------------------------------------------
__global__ __launch_bounds__(256) void k_gemm_feat(
    const __bf16* __restrict__ A, const __bf16* __restrict__ Bt,
    __bf16* __restrict__ F)
{
    int wave = threadIdx.x >> 5;
    int lane = threadIdx.x & 31;
    int tile = blockIdx.x * 8 + wave;   // 0..32767
    int mt = tile >> 4, nt = tile & 15;
    int hi = lane >> 4, l = lane & 15;

    const __bf16* arow = A  + (size_t)(mt * 16 + l) * 192;
    const __bf16* brow = Bt + (size_t)(nt * 16 + l) * 192;

    v8f c = {};
#pragma unroll
    for (int ks = 0; ks < 6; ++ks) {
        int kb = ks * 32;
        // A-frag (16x32 bf16): lane m=l, e<8: K=kb+8*hi+e ; e>=8: K=kb+16+8*hi+(e-8)
        v16bf a = cat16(ld8(arow + kb + 8 * hi), ld8(arow + kb + 16 + 8 * hi));
        // B-frag (32x16 bf16): lane n=l, e<8: K=kb+8*hi+e ; e>=8: K=kb+16+8*hi+(e-8)
        v16bf b = cat16(ld8(brow + kb + 8 * hi), ld8(brow + kb + 16 + 8 * hi));
        c = wmma_bf16(a, b, c);
    }
#pragma unroll
    for (int j = 0; j < 8; ++j)
        F[(size_t)(mt * 16 + 8 * hi + j) * 256 + nt * 16 + l] = (__bf16)c[j];
}

// ---------------------------------------------------------------------------
// Kernel 2: fused similarity -> online softmax over r -> label aggregation.
// Block = 4 waves, one 16-target tile; each wave covers 768 of 3072 ref rows,
// partial (m, s, O) merged in LDS (flash-attention combine).
// ---------------------------------------------------------------------------
__global__ __launch_bounds__(128) void k_attn(
    const __bf16* __restrict__ F, const __bf16* __restrict__ LT,
    float* __restrict__ out)
{
    int n  = blockIdx.x >> 6;
    int t0 = (blockIdx.x & 63) << 4;
    int w    = threadIdx.x >> 5;
    int lane = threadIdx.x & 31;
    int hi = lane >> 4, l = lane & 15;

    const __bf16* fbase = F + (size_t)n * 4096 * 256;

    // Preload target B-frags (loop-invariant): e<8: tgt[t0+l][kb+8*hi+e] ; e>=8: tgt[t0+l][kb+16+8*hi+(e-8)]
    v16bf Bf[8];
    const __bf16* trow = fbase + (size_t)(3072 + t0 + l) * 256;
#pragma unroll
    for (int ks = 0; ks < 8; ++ks) {
        const __bf16* pb = trow + ks * 32 + 8 * hi;
        Bf[ks] = cat16(ld8(pb), ld8(pb + 16));
    }

    float m = -INFINITY, s = 0.0f;
    v8f O = {};

    int rbeg = w * 768, rend = rbeg + 768;
    for (int rb = rbeg; rb < rend; rb += 32) {
        // S[r][t] for 32 r's: two 16x16 tiles, K = 256 over 8 wmma steps.
        v8f S0 = {}, S1 = {};
        const __bf16* a0row = fbase + (size_t)(rb + l) * 256;
        const __bf16* a1row = a0row + 16 * 256;
#pragma unroll
        for (int ks = 0; ks < 8; ++ks) {
            int kb = ks * 32;
            v16bf a0 = cat16(ld8(a0row + kb + 8 * hi), ld8(a0row + kb + 16 + 8 * hi));
            S0 = wmma_bf16(a0, Bf[ks], S0);
            v16bf a1 = cat16(ld8(a1row + kb + 8 * hi), ld8(a1row + kb + 16 + 8 * hi));
            S1 = wmma_bf16(a1, Bf[ks], S1);
        }
        // Per-target (lane column) chunk max over the 32 r values.
        float cm = S0[0];
#pragma unroll
        for (int j = 1; j < 8; ++j) cm = fmaxf(cm, S0[j]);
#pragma unroll
        for (int j = 0; j < 8; ++j) cm = fmaxf(cm, S1[j]);
        cm = fmaxf(cm, __shfl_xor(cm, 16, 32));
        float mnew  = fmaxf(m, cm);
        float alpha = __expf(m - mnew);

        float p0[8], p1[8], csum = 0.0f;
#pragma unroll
        for (int j = 0; j < 8; ++j) {
            p0[j] = __expf(S0[j] - mnew);
            p1[j] = __expf(S1[j] - mnew);
            csum += p0[j] + p1[j];
        }
        csum += __shfl_xor(csum, 16, 32);
        s = s * alpha + csum;
        m = mnew;

        // Rescale O (O rows are t = 8*hi+j; alpha lives at lane t).
#pragma unroll
        for (int j = 0; j < 8; ++j) O[j] *= __shfl(alpha, 8 * hi + j, 32);

        // S-tile C/D layout == A-frag layout for P^T x L: no lane exchange.
        v16bf P;
#pragma unroll
        for (int j = 0; j < 8; ++j) { P[j] = (__bf16)p0[j]; P[8 + j] = (__bf16)p1[j]; }

        // Label B-frag: e<8: LT[n][l][rb+8*hi+e] ; e>=8: LT[n][l][rb+16+8*hi+(e-8)]
        const __bf16* lrow = LT + ((size_t)(n * 16 + l)) * 3072 + rb + 8 * hi;
        v16bf Lf = cat16(ld8(lrow), ld8(lrow + 16));
        O = wmma_bf16(P, Lf, O);
    }

    // Cross-wave flash combine in LDS.
    __shared__ float sm[4][16];
    __shared__ float ss[4][16];
    __shared__ float sO[4][16][16];
    if (hi == 0) { sm[w][l] = m; ss[w][l] = s; }
#pragma unroll
    for (int j = 0; j < 8; ++j) sO[w][8 * hi + j][l] = O[j];
    __syncthreads();

    if (w == 0) {
        float* orow = out + ((size_t)n * 1024 + t0) * 16;
#pragma unroll
        for (int j = 0; j < 8; ++j) {
            int t = 8 * hi + j;
            float M = sm[0][t];
            for (int q = 1; q < 4; ++q) M = fmaxf(M, sm[q][t]);
            float S = 0.0f, V = 0.0f;
            for (int q = 0; q < 4; ++q) {
                float e = __expf(sm[q][t] - M);
                S += ss[q][t] * e;
                V += sO[q][t][l] * e;
            }
            orow[t * 16 + l] = V / S;
        }
    }
}

// ---------------------------------------------------------------------------
// Host entry
// ---------------------------------------------------------------------------
extern "C" void kernel_launch(void* const* d_in, const int* in_sizes, int n_in,
                              void* d_out, int out_size, void* d_ws, size_t ws_size,
                              hipStream_t stream)
{
    const float* ref = (const float*)d_in[0];   // [8,3,256,256,3]
    const float* tgt = (const float*)d_in[1];   // [8,1,256,256,3]
    const float* lab = (const float*)d_in[2];   // [8,3,32,32,16]
    const float* wfe = (const float*)d_in[3];   // [8,8,3,256]
    float* out = (float*)d_out;                  // [8,1,32,32,16]

    char* ws = (char*)d_ws;
    __bf16* Abuf = (__bf16*)(ws);                                   // 32768*192 bf16 = 12.0 MiB
    __bf16* Btb  = (__bf16*)(ws + 12582912);                        // 256*192         96 KiB
    __bf16* Feat = (__bf16*)(ws + 12582912 + 98304);                // 32768*256      16.0 MiB
    __bf16* LTb  = (__bf16*)(ws + 12582912 + 98304 + 16777216);     // 8*16*3072     768 KiB

    k_patchify <<<32768, 192, 0, stream>>>(ref, tgt, Abuf);
    k_wcvt     <<<192,   256, 0, stream>>>(wfe, Btb);
    k_lcvt     <<<1536,  256, 0, stream>>>(lab, LTb);
    k_gemm_feat<<<4096,  256, 0, stream>>>(Abuf, Btb, Feat);
    k_attn     <<<512,   128, 0, stream>>>(Feat, LTb, out);
}
